// DiffSchNet_66116726554888
// MI455X (gfx1250) — matrix operand order repaired and checked by reading, not compiled, not run on hardware
//
#include <hip/hip_runtime.h>
#include <cstdint>
#include <cstddef>

typedef __attribute__((ext_vector_type(16))) _Float16 v16h;
typedef __attribute__((ext_vector_type(8)))  _Float16 v8h;
typedef __attribute__((ext_vector_type(8)))  float    v8f;
typedef _Float16 f16;

#define B_    256
#define NEL   32
#define ETOT  1120
#define ROWS  (B_*ETOT)      // 286720 edge rows
#define DIFF  224            // 7*DFEAT
#define HIDP  192            // HID_W=169 padded to 192 (6 k-steps of 32)
#define KER   128
#define EMB   256
#define ME    (B_*NEL)       // 8192 electron rows

// ---------------- WMMA operand loaders ----------------
// A-operand 16x32 f16 tile (M x K). Lane L(0-15): row M=L, K 0..15 contiguous;
// lane L+16: row L, K 16..31 (same k order as load_tileB).
__device__ __forceinline__ v16h load_tileA(const f16* base, int rowStride, int lane) {
  int row = lane & 15, hf = lane >> 4;
  const f16* p = base + (size_t)row * rowStride + hf * 16;
  return *(const v16h*)p;
}
// B-operand 32x16 f16 tile (K x N), from weights stored transposed [N][K].
// Lane L(0-15): col N=L, K 0..15 contiguous; lane L+16: col L, K 16..31.
__device__ __forceinline__ v16h load_tileB(const f16* baseT, int rowStride, int lane) {
  int ncol = lane & 15, hf = lane >> 4;
  const f16* p = baseT + (size_t)ncol * rowStride + hf * 16;
  return *(const v16h*)p;   // 32B contiguous, 32B aligned by construction
}

// ---------------- edge index decode (matches reference tables) ----------------
__device__ __forceinline__ void decode_edge(int e, int& s, int& r, int& et) {
  if (e < 480) {            // same-spin: per block sigma, row-major skip diagonal
    et = 0; int sg = e / 240, k = e % 240;
    int i = k / 15, jj = k % 15; int j = jj + (jj >= i ? 1 : 0);
    s = sg * 16 + i; r = sg * 16 + j;
  } else if (e < 992) {     // anti-spin: [up->down | down->up]
    et = 1; int k = e - 480;
    if (k < 256) { s = k >> 4; r = 16 + (k & 15); }
    else         { s = 16 + (k & 15); r = (k - 256) >> 4; }
  } else {                  // nuc->elec
    et = 2; int k = e - 992; s = k >> 5; r = k & 31;
  }
}
__device__ __forceinline__ int sender_of(int e, int et) {
  if (et == 0) { int sg = e / 240, k = e % 240; return sg * 16 + k / 15; }
  int k = e - 480;
  return (k < 256) ? (k >> 4) : (16 + (k & 15));
}

// ---------------- weight prep: pad + transpose to B layout, f16 ----------------
__global__ void prep_kernel(const float* __restrict__ wW1, const float* __restrict__ wb1,
                            const float* __restrict__ wW2, const float* __restrict__ hW,
                            const float* __restrict__ gW,  const float* __restrict__ Xemb,
                            f16* W1t, float* B1, f16* W2t, f16* HWt, f16* GWt,
                            float* elecF, f16* elecH)
{
  int task = blockIdx.y;
  int i = blockIdx.x * blockDim.x + threadIdx.x;
  if (task == 0) {                       // W1t[lt][n<192][k<224] = wW1[lt][k][n<169]
    if (i < 9 * HIDP * DIFF) {
      int k = i % DIFF, n = (i / DIFF) % HIDP, lt = i / (DIFF * HIDP);
      float v = (n < 169) ? wW1[((size_t)lt * DIFF + k) * 169 + n] : 0.f;
      W1t[((size_t)lt * HIDP + n) * DIFF + k] = (f16)v;
    }
  } else if (task == 1) {                // padded bias (f32)
    if (i < 9 * HIDP) {
      int n = i % HIDP, lt = i / HIDP;
      B1[i] = (n < 169) ? wb1[lt * 169 + n] : 0.f;
    }
  } else if (task == 2) {                // W2t[lt][n<128][k<192] = wW2[lt][k<169][n]
    if (i < 9 * KER * HIDP) {
      int k = i % HIDP, n = (i / HIDP) % KER, lt = i / (HIDP * KER);
      float v = (k < 169) ? wW2[((size_t)lt * 169 + k) * KER + n] : 0.f;
      W2t[((size_t)lt * KER + n) * HIDP + k] = (f16)v;
    }
  } else if (task == 3) {                // HWt[l][c][k]; c = sigma*128 + kcol
    if (i < 2 * 256 * 256) {
      int k = i % 256, c = (i / 256) % 256, l = i / (256 * 256);
      HWt[((size_t)l * 256 + c) * 256 + k] =
          (f16)hW[(((size_t)l * 2 + (c >> 7)) * 256 + k) * 128 + (c & 127)];
    }
  } else if (task == 4) {                // GWt[l][n][kk]; kk = slot*128 + k
    if (i < 3 * EMB * 384) {
      int kk = i % 384, n = (i / 384) % EMB, l = i / (384 * EMB);
      GWt[((size_t)l * EMB + n) * 384 + kk] =
          (f16)gW[(((size_t)l * 3 + (kk >> 7)) * 128 + (kk & 127)) * EMB + n];
    }
  } else {                               // elec init = broadcast X_emb row
    if (i < ME * EMB) {
      float v = Xemb[i & (EMB - 1)];
      elecF[i] = v; elecH[i] = (f16)v;
    }
  }
}

// ---------------- edge features: one wave per edge, lane = gaussian ----------------
__global__ void edge_feat_kernel(const float* __restrict__ rs,
                                 const float* __restrict__ coords,
                                 f16* __restrict__ dexp)
{
  int wave = blockIdx.x * (blockDim.x >> 5) + (threadIdx.x >> 5);
  int lane = threadIdx.x & 31;
  if (wave >= ROWS) return;
  int b = wave / ETOT, e = wave % ETOT;
  int s, r, et; decode_edge(e, s, r, et);
  const float* pr = rs + ((size_t)b * NEL + r) * 3;
  const float* ps = (et == 2) ? (coords + s * 3) : (rs + ((size_t)b * NEL + s) * 3);
  float d0 = ps[0] - pr[0], d1 = ps[1] - pr[1], d2 = ps[2] - pr[2];
  float q = (1.f + 2.f * lane) * (1.f / 64.f);
  float mu = 10.f * q * q;
  float invsig = 7.f / (1.f + 10.f * q);
  f16* out = dexp + (size_t)wave * DIFF + lane;
#pragma unroll
  for (int g = 0; g < 7; ++g) {
    float dc = (g >> 1 == 0) ? d0 : ((g >> 1 == 1) ? d1 : d2);
    float dd = (g < 6) ? ((g & 1) ? fmaxf(-dc, 0.f) : fmaxf(dc, 0.f)) : d2;
    float env = dd * dd * __expf(-dd);
    float t = (dd - mu) * invsig;
    out[g * 32] = (f16)(env * __expf(-t * t));
  }
}

// ---------------- fused edge MLP: GEMM1 + silu + GEMM2 + modulation ----------------
__global__ __launch_bounds__(128) void gemm12_kernel(
    const f16* __restrict__ dexp, const f16* __restrict__ W1t,
    const float* __restrict__ B1, const f16* __restrict__ W2t,
    const f16* __restrict__ HS,  const float* __restrict__ h0,
    const float* __restrict__ Yw, f16* __restrict__ weh, int l)
{
  __shared__ f16 smem[4][16][HIDP];
  int wid = threadIdx.x >> 5, lane = threadIdx.x & 31;
  int t = blockIdx.x * 4 + wid;                 // 16-row tile id, < 17920
  int colc = lane & 15, hf = lane >> 4;
  int e0 = (t * 16) % ETOT;
  int et = (e0 < 480) ? 0 : (e0 < 992 ? 1 : 2); // uniform per tile (segments %16==0)
  int lt = l * 3 + et;
  const f16* A   = dexp + (size_t)t * 16 * DIFF;
  const f16* Bt1 = W1t + (size_t)lt * HIDP * DIFF;

  v8f acc[12];
#pragma unroll
  for (int n = 0; n < 12; ++n)
#pragma unroll
    for (int i = 0; i < 8; ++i) acc[n][i] = 0.f;

  // software-pipelined: prefetch next A (k-step) and next B tile ahead of WMMA
  v16h a = load_tileA(A, DIFF, lane);
  for (int kk = 0; kk < 7; ++kk) {              // K = 224
    v16h anx = a;
    if (kk < 6) anx = load_tileA(A + (kk + 1) * 32, DIFF, lane);
    v16h bm = load_tileB(Bt1 + kk * 32, DIFF, lane);
#pragma unroll
    for (int n = 0; n < 12; ++n) {
      v16h bnx = bm;
      if (n < 11) bnx = load_tileB(Bt1 + (size_t)(n + 1) * 16 * DIFF + kk * 32, DIFF, lane);
      acc[n] = __builtin_amdgcn_wmma_f32_16x16x32_f16(false, a, false, bm,
                                                      (short)0, acc[n], false, false);
      bm = bnx;
    }
    a = anx;
  }
  // bias + silu, stage h tile (16x192 f16) in LDS for re-layout
  const float* bias = B1 + lt * HIDP;
#pragma unroll
  for (int n = 0; n < 12; ++n) {
    float bv = bias[n * 16 + colc];
#pragma unroll
    for (int rr = 0; rr < 8; ++rr) {
      int rowt = rr + 8 * hf;
      float x = acc[n][rr] + bv;
      smem[wid][rowt][n * 16 + colc] = (f16)(x / (1.f + __expf(-x)));
    }
  }
  // GEMM2: h (from LDS, A layout) x W2t -> 16x128
  v8f acc2[8];
#pragma unroll
  for (int n = 0; n < 8; ++n)
#pragma unroll
    for (int i = 0; i < 8; ++i) acc2[n][i] = 0.f;
  const f16* Bt2 = W2t + (size_t)lt * KER * HIDP;
  const f16* As = &smem[wid][0][0];
  v16h a2 = load_tileA(As, HIDP, lane);
  for (int kk = 0; kk < 6; ++kk) {              // K = 192
    v16h anx = a2;
    if (kk < 5) anx = load_tileA(As + (kk + 1) * 32, HIDP, lane);
    v16h bm = load_tileB(Bt2 + kk * 32, HIDP, lane);
#pragma unroll
    for (int n = 0; n < 8; ++n) {
      v16h bnx = bm;
      if (n < 7) bnx = load_tileB(Bt2 + (size_t)(n + 1) * 16 * HIDP + kk * 32, HIDP, lane);
      acc2[n] = __builtin_amdgcn_wmma_f32_16x16x32_f16(false, a2, false, bm,
                                                       (short)0, acc2[n], false, false);
      bm = bnx;
    }
    a2 = anx;
  }
  // modulate by sender embedding and store weh (f16)
#pragma unroll
  for (int rr = 0; rr < 8; ++rr) {
    int rowt = rr + 8 * hf;
    int ge = t * 16 + rowt;
    int b = ge / ETOT, e = ge % ETOT;
    const f16* hsrow = nullptr; const float* frow = nullptr;
    if (et == 2)      frow = Yw + ((e - 992) >> 5) * KER;          // nuc[NE_S]
    else if (l == 0)  frow = h0 + et * KER;                        // h0_emb[spin]
    else              hsrow = HS + ((size_t)(b * NEL + sender_of(e, et))) * 256 + et * KER;
    f16* wout = weh + (size_t)ge * KER;
#pragma unroll
    for (int n = 0; n < 8; ++n) {
      int col = n * 16 + colc;
      float m = hsrow ? (float)hsrow[col] : frow[col];
      wout[col] = (f16)(acc2[n][rr] * m);
    }
  }
}

// ---------------- HS = elec @ [h_W0 | h_W1]  (l >= 1) ----------------
__global__ __launch_bounds__(128) void hs_kernel(const f16* __restrict__ elecH,
                                                 const f16* __restrict__ HWt,
                                                 f16* __restrict__ HS, int l)
{
  int wid = threadIdx.x >> 5, lane = threadIdx.x & 31;
  int t = blockIdx.x * 4 + wid;                 // < 512
  int colc = lane & 15, hf = lane >> 4;
  const f16* A  = elecH + (size_t)t * 16 * EMB;
  const f16* Bt = HWt + (size_t)(l - 1) * 256 * 256;
  v8f acc[16];
#pragma unroll
  for (int n = 0; n < 16; ++n)
#pragma unroll
    for (int i = 0; i < 8; ++i) acc[n][i] = 0.f;
  v16h a = load_tileA(A, EMB, lane);
  for (int kk = 0; kk < 8; ++kk) {              // K = 256
    v16h anx = a;
    if (kk < 7) anx = load_tileA(A + (kk + 1) * 32, EMB, lane);
    v16h bm = load_tileB(Bt + kk * 32, 256, lane);
#pragma unroll
    for (int n = 0; n < 16; ++n) {
      v16h bnx = bm;
      if (n < 15) bnx = load_tileB(Bt + (size_t)(n + 1) * 16 * 256 + kk * 32, 256, lane);
      acc[n] = __builtin_amdgcn_wmma_f32_16x16x32_f16(false, a, false, bm,
                                                      (short)0, acc[n], false, false);
      bm = bnx;
    }
    a = anx;
  }
#pragma unroll
  for (int n = 0; n < 16; ++n)
#pragma unroll
    for (int rr = 0; rr < 8; ++rr)
      HS[((size_t)(t * 16 + rr + 8 * hf)) * 256 + n * 16 + colc] = (f16)acc[n][rr];
}

// ---------------- deterministic gather-scatter: Z[b][r][slot*128+c] ----------------
__global__ void gather_kernel(const f16* __restrict__ weh, f16* __restrict__ Zb)
{
  int idx = blockIdx.x * blockDim.x + threadIdx.x;  // < ME*384
  int col = idx % 384;
  int rr  = (idx / 384) % NEL;
  int b   = idx / (384 * NEL);
  int slot = col >> 7, c = col & 127;
  const f16* W = weh + (size_t)b * ETOT * KER + c;
  float sum = 0.f;
  if (slot == 0) {                                   // same-spin receivers
    int sg = rr >> 4, j = rr & 15, base = sg * 240;
#pragma unroll
    for (int i = 0; i < 16; ++i) {
      if (i == j) continue;
      int jj = j - (j > i ? 1 : 0);
      sum += (float)W[(size_t)(base + i * 15 + jj) * KER];
    }
  } else if (slot == 1) {                            // anti-spin receivers
    if (rr < 16) {
#pragma unroll
      for (int j2 = 0; j2 < 16; ++j2) sum += (float)W[(size_t)(736 + rr * 16 + j2) * KER];
    } else {
      int j = rr - 16;
#pragma unroll
      for (int i = 0; i < 16; ++i) sum += (float)W[(size_t)(480 + i * 16 + j) * KER];
    }
  } else {                                           // nuc->elec receivers
#pragma unroll
    for (int n = 0; n < 4; ++n) sum += (float)W[(size_t)(992 + n * 32 + rr) * KER];
  }
  Zb[idx] = (f16)sum;
}

// ---------------- readout: Z @ [g0;g1;g2], accumulate into elec ----------------
__global__ __launch_bounds__(128) void gemm3_kernel(const f16* __restrict__ Zb,
                                                    const f16* __restrict__ GWt,
                                                    float* __restrict__ elecF,
                                                    f16* __restrict__ elecH, int l)
{
  int wid = threadIdx.x >> 5, lane = threadIdx.x & 31;
  int t = blockIdx.x * 4 + wid;                 // < 512
  int colc = lane & 15, hf = lane >> 4;
  const f16* A  = Zb + (size_t)t * 16 * 384;
  const f16* Bt = GWt + (size_t)l * EMB * 384;
  v8f acc[16];
#pragma unroll
  for (int n = 0; n < 16; ++n)
#pragma unroll
    for (int i = 0; i < 8; ++i) acc[n][i] = 0.f;
  v16h a = load_tileA(A, 384, lane);
  for (int kk = 0; kk < 12; ++kk) {             // K = 384
    v16h anx = a;
    if (kk < 11) anx = load_tileA(A + (kk + 1) * 32, 384, lane);
    v16h bm = load_tileB(Bt + kk * 32, 384, lane);
#pragma unroll
    for (int n = 0; n < 16; ++n) {
      v16h bnx = bm;
      if (n < 15) bnx = load_tileB(Bt + (size_t)(n + 1) * 16 * 384 + kk * 32, 384, lane);
      acc[n] = __builtin_amdgcn_wmma_f32_16x16x32_f16(false, a, false, bm,
                                                      (short)0, acc[n], false, false);
      bm = bnx;
    }
    a = anx;
  }
#pragma unroll
  for (int n = 0; n < 16; ++n)
#pragma unroll
    for (int rr = 0; rr < 8; ++rr) {
      size_t idx = ((size_t)(t * 16 + rr + 8 * hf)) * EMB + n * 16 + colc;
      float v = elecF[idx] + acc[n][rr];
      elecF[idx] = v;
      elecH[idx] = (f16)v;
    }
}

// ---------------- host launch ----------------
extern "C" void kernel_launch(void* const* d_in, const int* in_sizes, int n_in,
                              void* d_out, int out_size, void* d_ws, size_t ws_size,
                              hipStream_t stream)
{
  const float* rs     = (const float*)d_in[0];
  const float* coords = (const float*)d_in[1];
  const float* Xemb   = (const float*)d_in[2];
  const float* Yw     = (const float*)d_in[3];
  const float* wW1    = (const float*)d_in[4];
  const float* wb1    = (const float*)d_in[5];
  const float* wW2    = (const float*)d_in[6];
  const float* h0     = (const float*)d_in[7];
  const float* hW     = (const float*)d_in[8];
  const float* gW     = (const float*)d_in[9];
  (void)in_sizes; (void)n_in; (void)out_size; (void)ws_size;

  char* ws = (char*)d_ws;
  size_t off = 0;
  auto alloc = [&](size_t bytes) -> char* {
    char* p = ws + off;
    off = (off + bytes + 255) & ~(size_t)255;
    return p;
  };
  f16*   dexp  = (f16*)  alloc((size_t)ROWS * DIFF * 2);   // 128.5 MB (L2-resident)
  f16*   weh   = (f16*)  alloc((size_t)ROWS * KER  * 2);   // 73.4 MB
  f16*   Zb    = (f16*)  alloc((size_t)ME * 384 * 2);
  f16*   HS    = (f16*)  alloc((size_t)ME * EMB * 2);
  f16*   elecH = (f16*)  alloc((size_t)ME * EMB * 2);
  f16*   W1t   = (f16*)  alloc((size_t)9 * HIDP * DIFF * 2);
  float* B1    = (float*)alloc((size_t)9 * HIDP * 4);
  f16*   W2t   = (f16*)  alloc((size_t)9 * KER * HIDP * 2);
  f16*   HWt   = (f16*)  alloc((size_t)2 * 256 * 256 * 2);
  f16*   GWt   = (f16*)  alloc((size_t)3 * EMB * 384 * 2);
  float* elecF = (float*)d_out;                            // output doubles as f32 state

  prep_kernel<<<dim3(8192, 6), 256, 0, stream>>>(wW1, wb1, wW2, hW, gW, Xemb,
                                                 W1t, B1, W2t, HWt, GWt, elecF, elecH);
  edge_feat_kernel<<<ROWS / 8, 256, 0, stream>>>(rs, coords, dexp);
  for (int l = 0; l < 3; ++l) {
    if (l > 0)
      hs_kernel<<<ME / 64, 128, 0, stream>>>(elecH, HWt, HS, l);
    gemm12_kernel<<<ROWS / 64, 128, 0, stream>>>(dexp, W1t, B1, W2t, HS, h0, Yw, weh, l);
    gather_kernel<<<ME * 384 / 256, 256, 0, stream>>>(weh, Zb);
    gemm3_kernel<<<ME / 64, 128, 0, stream>>>(Zb, GWt, elecF, elecH, l);
  }
}
